// GPT_37950331027559
// MI455X (gfx1250) — matrix operand order repaired and checked by reading, not compiled, not run on hardware
//
#include <hip/hip_runtime.h>

typedef __attribute__((ext_vector_type(16))) _Float16 v16h;
typedef __attribute__((ext_vector_type(8)))  float    v8f;

#define B_  4
#define T_  1024
#define E_  1024
#define H_  16
#define HS_ 64
#define L_  8
#define F_  4096
#define V_  50257
#define M_  (B_*T_)

// 32-bit LDS byte offset for async-to-LDS instructions (generic -> AS3 cast)
__device__ __forceinline__ unsigned lds_addr32(const void* p) {
  return (unsigned)(unsigned long long)(const __attribute__((address_space(3))) char*)p;
}

// ---------------- embedding: x = tok_emb[idx] + pos_emb ----------------
__global__ __launch_bounds__(256) void embed_kernel(const long long* __restrict__ idx,
    const float* __restrict__ tok, const float* __restrict__ pos, float* __restrict__ x)
{
  int r = blockIdx.x;
  int t = r % T_;
  long long v = idx[r];
  const float* tr = tok + (size_t)v * E_;
  const float* pr = pos + (size_t)t * E_;
  float* xr = x + (size_t)r * E_;
  #pragma unroll
  for (int j = 0; j < E_/256; ++j) {
    int c = threadIdx.x + j * 256;
    xr[c] = tr[c] + pr[c];
  }
}

// ---------------- LayerNorm fp32 -> f16 (one wave per row) ----------------
__global__ __launch_bounds__(256) void ln_f16_kernel(const float* __restrict__ xin,
    const float* __restrict__ g, const float* __restrict__ b, _Float16* __restrict__ out)
{
  int wid  = threadIdx.x >> 5;
  int lane = threadIdx.x & 31;
  int row  = blockIdx.x * 8 + wid;
  const float* xr = xin + (size_t)row * E_;
  float s = 0.f, ss = 0.f;
  #pragma unroll
  for (int j = 0; j < E_/32; ++j) { float v = xr[lane + j*32]; s += v; ss += v*v; }
  #pragma unroll
  for (int m = 16; m >= 1; m >>= 1) { s += __shfl_xor(s, m, 32); ss += __shfl_xor(ss, m, 32); }
  float mu   = s  * (1.f/E_);
  float var  = ss * (1.f/E_) - mu*mu;
  float rstd = rsqrtf(var + 1e-5f);
  _Float16* orow = out + (size_t)row * E_;
  #pragma unroll
  for (int j = 0; j < E_/32; ++j) {
    int c = lane + j*32;
    orow[c] = (_Float16)(((xr[c]-mu)*rstd)*g[c] + b[c]);
  }
}

// ---------------- generic WMMA GEMM, async-LDS + double-buffered ----------------
// C[M,N] = act(A[M,K](f16) @ B[K,N](f32->f16) + bias) (+ resid), out fp32 or f16.
// grid.z applies per-z B offset (bzs elems) and C column offset (czs elems).
// BT=true: B element (k,n) is at Bg[n*ldb + k] (for x @ tok_emb^T).
template<bool BT>
__global__ __launch_bounds__(256) void gemm_wmma(const _Float16* __restrict__ A,
    const float* __restrict__ Bg, const float* __restrict__ bias,
    const float* __restrict__ resid, float* __restrict__ Cf, _Float16* __restrict__ Ch,
    int Mm, int Nn, int Kk, long ldb, long ldc, long bzs, long czs, int relu)
{
  __shared__ _Float16 As[2][128*32];   // 128x32 f16 tile, async-filled (16B chunks)
  __shared__ _Float16 Bs[2][32*66];    // 32x64 tile (fp32->f16), padded stride 66
  const int tid  = threadIdx.x;
  const int lane = tid & 31, wid = tid >> 5;
  const int wm = wid & 3, wn = wid >> 2;        // 4 x 2 waves
  const int bm0 = blockIdx.x * 128;
  const int bn0 = blockIdx.y * 64;
  Bg += (long)blockIdx.z * bzs;
  const long coff = (long)blockIdx.z * czs;
  const int halfsel = (lane >= 16) ? 1 : 0;
  const int l16 = lane & 15;

  const unsigned asbase0 = lds_addr32(&As[0][0]);
  const unsigned asbase1 = lds_addr32(&As[1][0]);

  v8f acc[2][2] = {{{},{}},{{},{}}};

  // async stage of A tile: 512 x 16B chunks, 2 per thread
  auto stageA = [&](int kb, int buf) {
    unsigned base = buf ? asbase1 : asbase0;
    #pragma unroll
    for (int j = 0; j < 2; ++j) {
      int c = tid + j*256;
      int row = c >> 2, part = c & 3;          // 4 x 16B chunks per 64B row
      const _Float16* gp = A + (size_t)(bm0 + row)*Kk + kb + part*8;
      unsigned lo = base + (unsigned)(row*32 + part*8) * 2u;
      asm volatile("global_load_async_to_lds_b128 %0, %1, off"
                   :: "v"(lo), "v"(gp) : "memory");
    }
  };
  // B tile: fp32 load -> f16 convert -> LDS
  auto stageB = [&](int kb, int buf) {
    #pragma unroll
    for (int j = 0; j < 8; ++j) {
      int e = tid + j*256;
      int k = e >> 6, n = e & 63;
      int gn = bn0 + n;
      float v = 0.f;
      if (gn < Nn) v = BT ? Bg[(size_t)gn*ldb + kb + k] : Bg[(size_t)(kb+k)*ldb + gn];
      Bs[buf][k*66 + n] = (_Float16)v;
    }
    if (kb + 64 < Kk) {       // stream hint one tile further ahead
      int gn = bn0 + (tid & 63);
      if (gn < Nn) {
        const float* pf = BT ? &Bg[(size_t)gn*ldb + kb + 64]
                             : &Bg[(size_t)(kb + 64 + (tid >> 6)*8)*ldb + gn];
        __builtin_prefetch(pf, 0, 1);
      }
    }
  };

  stageA(0, 0);
  stageB(0, 0);

  for (int kb = 0; kb < Kk; kb += 32) {
    const int buf = (kb >> 5) & 1;
    // our wave's async LDS writes done, then workgroup-wide visibility
    asm volatile("s_wait_asynccnt 0x0" ::: "memory");
    __syncthreads();
    if (kb + 32 < Kk) {                 // pipeline next tile into other buffer
      stageA(kb + 32, buf ^ 1);
      stageB(kb + 32, buf ^ 1);
    }
    v16h af[2], bf[2];
    #pragma unroll
    for (int mi = 0; mi < 2; ++mi) {
      int m = wm*32 + mi*16 + l16;
      #pragma unroll
      for (int e = 0; e < 16; ++e) {
        int k = e + ((e >= 8) ? 8 : 0) + halfsel*8;   // 16-bit A 16x32 layout
        af[mi][e] = As[buf][m*32 + k];
      }
    }
    #pragma unroll
    for (int ni = 0; ni < 2; ++ni) {
      int n = wn*32 + ni*16 + l16;
      #pragma unroll
      for (int e = 0; e < 16; ++e) {
        int k = e + ((e >= 8) ? 8 : 0) + halfsel*8;   // B 32x16 layout
        bf[ni][e] = Bs[buf][k*66 + n];
      }
    }
    #pragma unroll
    for (int mi = 0; mi < 2; ++mi)
      #pragma unroll
      for (int ni = 0; ni < 2; ++ni)
        acc[mi][ni] = __builtin_amdgcn_wmma_f32_16x16x32_f16(
            false, af[mi], false, bf[ni], (short)0, acc[mi][ni], false, false);
    // no trailing barrier: next iteration's top barrier orders buffer reuse
  }

  #pragma unroll
  for (int mi = 0; mi < 2; ++mi)
    #pragma unroll
    for (int ni = 0; ni < 2; ++ni)
      #pragma unroll
      for (int r = 0; r < 8; ++r) {
        int grow = bm0 + wm*32 + mi*16 + r + halfsel*8;
        int gcol = bn0 + wn*32 + ni*16 + l16;
        if (gcol < Nn) {
          float f = acc[mi][ni][r];
          if (bias) f += bias[gcol];
          if (relu) f = fmaxf(f, 0.f);
          size_t ci = (size_t)grow*ldc + coff + gcol;
          if (resid) f += resid[ci];
          if (Cf) Cf[ci] = f; else Ch[ci] = (_Float16)f;
        }
      }
}

// ---------------- flash-style causal attention ----------------
// one wave per (b,h,16 query rows); Q,K,V,O stored [B,T,H,HS] (row = b*T+t, col = h*HS+d)
__global__ __launch_bounds__(32) void attn_kernel(const _Float16* __restrict__ Q,
    const _Float16* __restrict__ K, const _Float16* __restrict__ V, _Float16* __restrict__ O)
{
  __shared__ _Float16 P[16*32];
  const int lane = threadIdx.x;
  const int l16 = lane & 15;
  const int halfsel = (lane >= 16) ? 1 : 0;
  const int qbase = blockIdx.x * 16;
  const int bh = blockIdx.y;
  const int b = bh / H_, h = bh % H_;
  const size_t bT = (size_t)b * T_;
  const int hHS = h * HS_;

  v16h qf[2];  // Q tile 16x64 as two 16x32 A-fragments
  const _Float16* Qr = Q + (bT + qbase + l16) * E_ + hHS;
  #pragma unroll
  for (int f = 0; f < 2; ++f)
    #pragma unroll
    for (int e = 0; e < 16; ++e)
      qf[f][e] = Qr[f*32 + e + ((e >= 8) ? 8 : 0) + halfsel*8];

  v8f of[4] = {{},{},{},{}};
  float mi[8], li[8];
  #pragma unroll
  for (int r = 0; r < 8; ++r) { mi[r] = -3.0e38f; li[r] = 0.f; }

  const int smax = qbase + 15;
  for (int s0 = 0; s0 <= smax; s0 += 32) {
    v8f sf[2] = {{},{}};
    #pragma unroll
    for (int c = 0; c < 2; ++c) {                 // S = Q @ K^T, two 16-col chunks
      const _Float16* Kr = K + (bT + s0 + c*16 + l16) * E_ + hHS;
      #pragma unroll
      for (int f = 0; f < 2; ++f) {
        v16h bfr;
        #pragma unroll
        for (int e = 0; e < 16; ++e)
          bfr[e] = Kr[f*32 + e + ((e >= 8) ? 8 : 0) + halfsel*8];
        sf[c] = __builtin_amdgcn_wmma_f32_16x16x32_f16(
            false, qf[f], false, bfr, (short)0, sf[c], false, false);
      }
    }
    float p0[8], p1[8], alpha[8];
    #pragma unroll
    for (int r = 0; r < 8; ++r) {                 // online softmax, per-row over lanes
      int qi = qbase + r + halfsel*8;
      float a0 = sf[0][r] * 0.125f;               // HS^-0.5
      float a1 = sf[1][r] * 0.125f;
      if (s0 + l16 > qi)      a0 = -3.0e38f;      // causal mask
      if (s0 + 16 + l16 > qi) a1 = -3.0e38f;
      float rm = fmaxf(a0, a1);
      rm = fmaxf(rm, __shfl_xor(rm, 1, 32));
      rm = fmaxf(rm, __shfl_xor(rm, 2, 32));
      rm = fmaxf(rm, __shfl_xor(rm, 4, 32));
      rm = fmaxf(rm, __shfl_xor(rm, 8, 32));
      float mn = fmaxf(mi[r], rm);
      alpha[r] = __expf(mi[r] - mn);
      p0[r] = __expf(a0 - mn);
      p1[r] = __expf(a1 - mn);
      float rs = p0[r] + p1[r];
      rs += __shfl_xor(rs, 1, 32);
      rs += __shfl_xor(rs, 2, 32);
      rs += __shfl_xor(rs, 4, 32);
      rs += __shfl_xor(rs, 8, 32);
      li[r] = li[r] * alpha[r] + rs;
      mi[r] = mn;
    }
    #pragma unroll
    for (int t = 0; t < 4; ++t)
      #pragma unroll
      for (int r = 0; r < 8; ++r)
        of[t][r] *= alpha[r];
    #pragma unroll
    for (int r = 0; r < 8; ++r) {                 // P tile 16x32 to LDS (C->A relayout)
      int rr = r + halfsel*8;
      P[rr*32 + l16]      = (_Float16)p0[r];
      P[rr*32 + 16 + l16] = (_Float16)p1[r];
    }
    __syncthreads();
    v16h pf;
    #pragma unroll
    for (int e = 0; e < 16; ++e)
      pf[e] = P[l16*32 + e + ((e >= 8) ? 8 : 0) + halfsel*8];
    #pragma unroll
    for (int t = 0; t < 4; ++t) {                 // O += P @ V (4 d-tiles of 16)
      v16h bfr;
      #pragma unroll
      for (int e = 0; e < 16; ++e)
        bfr[e] = V[(bT + s0 + e + ((e >= 8) ? 8 : 0) + halfsel*8) * E_ + hHS + t*16 + l16];
      of[t] = __builtin_amdgcn_wmma_f32_16x16x32_f16(
          false, pf, false, bfr, (short)0, of[t], false, false);
    }
    __syncthreads();
  }
  #pragma unroll
  for (int t = 0; t < 4; ++t)
    #pragma unroll
    for (int r = 0; r < 8; ++r) {
      int row = qbase + r + halfsel*8;
      O[(bT + row) * E_ + hHS + t*16 + l16] = (_Float16)(of[t][r] / li[r]);
    }
}

// ---------------- host orchestration ----------------
extern "C" void kernel_launch(void* const* d_in, const int* in_sizes, int n_in,
                              void* d_out, int out_size, void* d_ws, size_t ws_size,
                              hipStream_t stream)
{
  (void)in_sizes; (void)n_in; (void)out_size; (void)ws_size;
  const long long* idx = (const long long*)d_in[0];
  const float* tok  = (const float*)d_in[1];
  const float* pos  = (const float*)d_in[2];
  const float* wq   = (const float*)d_in[3];
  const float* wk   = (const float*)d_in[4];
  const float* wv   = (const float*)d_in[5];
  const float* wo   = (const float*)d_in[6];
  const float* bo   = (const float*)d_in[7];
  const float* ln1g = (const float*)d_in[8];
  const float* ln1b = (const float*)d_in[9];
  const float* ln2g = (const float*)d_in[10];
  const float* ln2b = (const float*)d_in[11];
  const float* w1   = (const float*)d_in[12];
  const float* b1   = (const float*)d_in[13];
  const float* w2   = (const float*)d_in[14];
  const float* b2   = (const float*)d_in[15];
  const float* lnfg = (const float*)d_in[16];
  const float* lnfb = (const float*)d_in[17];

  char* ws = (char*)d_ws;                                // 88 MB total
  float*    x   = (float*)ws;                            // 16 MB  fp32 residual stream
  _Float16* hln = (_Float16*)(ws + (16u<<20));           //  8 MB  LN output (f16)
  _Float16* qb  = (_Float16*)(ws + (24u<<20));           //  8 MB
  _Float16* kb  = (_Float16*)(ws + (32u<<20));           //  8 MB
  _Float16* vb  = (_Float16*)(ws + (40u<<20));           //  8 MB
  _Float16* ob  = (_Float16*)(ws + (48u<<20));           //  8 MB  attention out (f16)
  _Float16* mid = (_Float16*)(ws + (56u<<20));           // 32 MB  MLP hidden (f16)

  embed_kernel<<<M_, 256, 0, stream>>>(idx, tok, pos, x);

  for (int l = 0; l < L_; ++l) {
    ln_f16_kernel<<<M_/8, 256, 0, stream>>>(x, ln1g + l*E_, ln1b + l*E_, hln);

    dim3 gq(M_/128, 1, H_);                    // per-head QKV GEMMs, N=HS
    const size_t wsz = (size_t)H_*E_*HS_;
    gemm_wmma<false><<<gq, 256, 0, stream>>>(hln, wq + (size_t)l*wsz, nullptr, nullptr,
        nullptr, qb, M_, HS_, E_, HS_, E_, (long)E_*HS_, HS_, 0);
    gemm_wmma<false><<<gq, 256, 0, stream>>>(hln, wk + (size_t)l*wsz, nullptr, nullptr,
        nullptr, kb, M_, HS_, E_, HS_, E_, (long)E_*HS_, HS_, 0);
    gemm_wmma<false><<<gq, 256, 0, stream>>>(hln, wv + (size_t)l*wsz, nullptr, nullptr,
        nullptr, vb, M_, HS_, E_, HS_, E_, (long)E_*HS_, HS_, 0);

    attn_kernel<<<dim3(T_/16, B_*H_), 32, 0, stream>>>(qb, kb, vb, ob);

    // x = x + attn @ wo + bo
    gemm_wmma<false><<<dim3(M_/128, E_/64), 256, 0, stream>>>(ob, wo + (size_t)l*E_*E_,
        bo + l*E_, x, x, nullptr, M_, E_, E_, E_, E_, 0, 0, 0);

    ln_f16_kernel<<<M_/8, 256, 0, stream>>>(x, ln2g + l*E_, ln2b + l*E_, hln);

    // mid = relu(h2 @ w1 + b1)
    gemm_wmma<false><<<dim3(M_/128, F_/64), 256, 0, stream>>>(hln, w1 + (size_t)l*E_*F_,
        b1 + (size_t)l*F_, nullptr, nullptr, mid, M_, F_, E_, F_, F_, 0, 0, 1);
    // x = x + mid @ w2 + b2
    gemm_wmma<false><<<dim3(M_/128, E_/64), 256, 0, stream>>>(mid, w2 + (size_t)l*F_*E_,
        b2 + l*E_, x, x, nullptr, M_, E_, F_, E_, E_, 0, 0, 0);
  }

  ln_f16_kernel<<<M_/8, 256, 0, stream>>>(x, lnfg, lnfb, hln);

  // logits = hln @ tok_emb^T  (BT mode)
  gemm_wmma<true><<<dim3(M_/128, (V_+63)/64), 256, 0, stream>>>(hln, tok, nullptr, nullptr,
      (float*)d_out, nullptr, M_, V_, E_, E_, V_, 0, 0, 0);
}
